// HawkesLSTMCell_9715216023707
// MI455X (gfx1250) — compile-verified, matrix operand order checked by reading, not run on hardware
//
#include <hip/hip_runtime.h>
#include <hip/hip_bf16.h>
#include <math.h>

// HawkesLSTMCell fused GEMM+gates for gfx1250 (MI455X, wave32, WMMA)
// g = [x|h] @ W + b  ->  sigmoid/tanh/softplus gate math, 4 outputs.
// v2: 2 M-tiles per wave (14 WMMA / 18 ds_load per k-tile), packed b32 B-staging.

typedef __attribute__((ext_vector_type(16))) __bf16 v16bf;
typedef __attribute__((ext_vector_type(8)))  __bf16 v8bf;
typedef __attribute__((ext_vector_type(4)))  __bf16 v4bf;
typedef __attribute__((ext_vector_type(2)))  __bf16 v2bf;
typedef __attribute__((ext_vector_type(8)))  float  v8f;

#define Bn 32768
#define Dn 512
#define Hn 512
#define Kn 1024
#define NG 7
#define BLOCK_M 128
#define BLOCK_H 32
#define KT 32
#define NKT (Kn / KT)          // 32 k-tiles
#define LDA 40                 // padded k-stride (elements) for sA rows
#define LDB 40                 // padded k-stride (elements) for sB cols
#define NCOL (NG * BLOCK_H)    // 224 staged W columns per block
#define NTHREADS 256           // 8 wave32 waves

__device__ __forceinline__ float sigm_f(float x) {
    return 1.0f / (1.0f + __expf(-x));
}
__device__ __forceinline__ float tanh_f(float x) {
    float a = fabsf(x);
    float e = __expf(2.0f * a);
    float t = 1.0f - 2.0f / (e + 1.0f);
    return copysignf(t, x);
}
__device__ __forceinline__ float softplus_f(float x) {
    // BETA = 1.0
    return (x > 20.0f) ? x : log1pf(__expf(x));
}

__global__ __launch_bounds__(NTHREADS, 1)
void hawkes_lstm_wmma_kernel(const float* __restrict__ x,
                             const float* __restrict__ ht,
                             const float* __restrict__ ct,
                             const float* __restrict__ ctar,
                             const float* __restrict__ W,
                             const float* __restrict__ bias,
                             float* __restrict__ out) {
    __shared__ __attribute__((aligned(16))) __bf16 sA[BLOCK_M * LDA];
    __shared__ __attribute__((aligned(16))) __bf16 sB[NCOL * LDB];

    const int tid   = threadIdx.x;
    const int lane  = tid & 31;
    const int wid   = tid >> 5;      // 0..7
    const int waveM = wid & 3;       // 4 waves along M, 32 rows (2 M-tiles) each
    const int waveN = wid >> 2;      // 2 waves along h, 16 cols each
    const int q     = lane >> 4;     // half-wave select (K-group)
    const int ln    = lane & 15;

    const int hBase   = blockIdx.x * BLOCK_H;
    const int rowBase = blockIdx.y * BLOCK_M;

    v8f acc[2][NG];
#pragma unroll
    for (int mt = 0; mt < 2; ++mt)
#pragma unroll
        for (int g = 0; g < NG; ++g)
#pragma unroll
            for (int v = 0; v < 8; ++v) acc[mt][g][v] = 0.0f;

    // Register staging buffers for the 2-stage software pipeline.
    float4 aReg[4];      // A tile: 1024 float4s / 256 thr = 4 each
    float2 bReg[7][2];   // B tile: 1792 (colpair x kpair) units / 256 thr = 7 each

    // ---- global -> register staging ----
    auto loadA = [&](int kt) {
        // K-tile lies entirely in x (kt<16) or h_t (kt>=16)
        const float* src = (kt < 16) ? x : ht;
        const int kbase  = (kt < 16) ? kt * KT : kt * KT - Dn;
#pragma unroll
        for (int i = 0; i < 4; ++i) {
            int f  = tid + i * NTHREADS;   // flat float4 index in [0,1024)
            int r  = f >> 3;               // row in tile
            int kq = f & 7;                // float4 within row
            aReg[i] = *(const float4*)(src + (size_t)(rowBase + r) * Dn + kbase + kq * 4);
        }
    };
    auto loadB = [&](int kt) {
        // unit = (column pair, k pair): two b64 row-coalesced loads
#pragma unroll
        for (int i = 0; i < 7; ++i) {
            int u   = tid + i * NTHREADS;  // [0,1792)
            int kp  = u / 112;             // k-pair 0..15
            int c2  = u % 112;             // col-pair 0..111
            int col = c2 * 2;              // even, never crosses gate chunk
            int g   = col >> 5;            // gate 0..6
            int cl  = col & 31;            // col within gate chunk
            int k   = kp * 2;
            const float* p = W + (size_t)(kt * KT + k) * (NG * Hn) + g * Hn + hBase + cl;
            bReg[i][0] = *(const float2*)(p);                 // row k:   cols (col, col+1)
            bReg[i][1] = *(const float2*)(p + NG * Hn);       // row k+1: cols (col, col+1)
        }
    };

    // ---- register -> LDS (with f32 -> bf16 conversion) ----
    auto storeA = [&]() {
#pragma unroll
        for (int i = 0; i < 4; ++i) {
            int f  = tid + i * NTHREADS;
            int r  = f >> 3;
            int kq = f & 7;
            v4bf pk;
            pk[0] = (__bf16)aReg[i].x; pk[1] = (__bf16)aReg[i].y;
            pk[2] = (__bf16)aReg[i].z; pk[3] = (__bf16)aReg[i].w;
            *(v4bf*)(&sA[r * LDA + kq * 4]) = pk;   // ds_store_b64
        }
    };
    auto storeB = [&]() {
#pragma unroll
        for (int i = 0; i < 7; ++i) {
            int u   = tid + i * NTHREADS;
            int kp  = u / 112;
            int c2  = u % 112;
            int col = c2 * 2;
            int k   = kp * 2;
            v2bf p0, p1;                       // pack adjacent k for same column
            p0[0] = (__bf16)bReg[i][0].x; p0[1] = (__bf16)bReg[i][1].x;
            p1[0] = (__bf16)bReg[i][0].y; p1[1] = (__bf16)bReg[i][1].y;
            *(v2bf*)(&sB[(col + 0) * LDB + k]) = p0;   // ds_store_b32
            *(v2bf*)(&sB[(col + 1) * LDB + k]) = p1;   // ds_store_b32
        }
    };

    // ---- WMMA compute on current LDS tile ----
    auto compute = [&]() {
        // A fragments (16x32 bf16): lanes 0-15 -> K {0..7,16..23}; lanes 16-31 -> {8..15,24..31}
        v16bf afrag[2];
#pragma unroll
        for (int mt = 0; mt < 2; ++mt) {
            const __bf16* pa = &sA[(waveM * 32 + mt * 16 + ln) * LDA + q * 8];
            ((v8bf*)&afrag[mt])[0] = *(const v8bf*)(pa);        // ds_load_b128
            ((v8bf*)&afrag[mt])[1] = *(const v8bf*)(pa + 16);   // ds_load_b128
        }
#pragma unroll
        for (int g = 0; g < NG; ++g) {
            v16bf bfrag;
            const __bf16* pb = &sB[(g * BLOCK_H + waveN * 16 + ln) * LDB + q * 8];
            ((v8bf*)&bfrag)[0] = *(const v8bf*)(pb);
            ((v8bf*)&bfrag)[1] = *(const v8bf*)(pb + 16);
            acc[0][g] = __builtin_amdgcn_wmma_f32_16x16x32_bf16(
                            false, afrag[0], false, bfrag,
                            (short)0, acc[0][g], false, false);
            acc[1][g] = __builtin_amdgcn_wmma_f32_16x16x32_bf16(
                            false, afrag[1], false, bfrag,
                            (short)0, acc[1][g], false, false);
        }
    };

    // ---- main K loop: 2-stage pipeline (global loads overlap compute) ----
    loadA(0);
    loadB(0);
#pragma unroll 1
    for (int kt = 0; kt < NKT; ++kt) {
        storeA();
        storeB();
        __syncthreads();
        if (kt + 1 < NKT) {          // issue next tile's global loads early
            loadA(kt + 1);
            loadB(kt + 1);
        }
        compute();
        __syncthreads();
    }

    // ---- fused epilogue: gate math per lane (gates share lane/slot layout) ----
    const int h = hBase + waveN * 16 + ln;
    float bb[NG];
#pragma unroll
    for (int g = 0; g < NG; ++g) bb[g] = bias[g * Hn + h];

    const size_t BH = (size_t)Bn * Hn;
#pragma unroll
    for (int mt = 0; mt < 2; ++mt) {
#pragma unroll
        for (int v = 0; v < 8; ++v) {
            const int row = rowBase + waveM * 32 + mt * 16 + v + 8 * q;
            const size_t o = (size_t)row * Hn + h;

            const float gi  = acc[mt][0][v] + bb[0];
            const float gf  = acc[mt][1][v] + bb[1];
            const float git = acc[mt][2][v] + bb[2];
            const float gft = acc[mt][3][v] + bb[3];
            const float go  = acc[mt][4][v] + bb[4];
            const float gz  = acc[mt][5][v] + bb[5];
            const float gd  = acc[mt][6][v] + bb[6];

            const float inpt   = sigm_f(gi);
            const float forget = sigm_f(gf);
            const float itar   = sigm_f(git);
            const float ftar   = sigm_f(gft);
            const float outp   = sigm_f(go);
            const float z      = tanh_f(gz);
            const float decay  = softplus_f(gd);

            const float cv  = ct[o];
            const float ctv = ctar[o];

            out[o]          = forget * cv + inpt * z;   // c_i
            out[BH + o]     = ftar * ctv + itar * z;    // c_target_new
            out[2 * BH + o] = outp;                     // output gate
            out[3 * BH + o] = decay;                    // decay
        }
    }
}

extern "C" void kernel_launch(void* const* d_in, const int* in_sizes, int n_in,
                              void* d_out, int out_size, void* d_ws, size_t ws_size,
                              hipStream_t stream) {
    (void)in_sizes; (void)n_in; (void)out_size; (void)d_ws; (void)ws_size;
    const float* x    = (const float*)d_in[0];
    const float* ht   = (const float*)d_in[1];
    const float* ct   = (const float*)d_in[2];
    const float* ctar = (const float*)d_in[3];
    const float* W    = (const float*)d_in[4];
    const float* bias = (const float*)d_in[5];
    float* out = (float*)d_out;

    dim3 grid(Hn / BLOCK_H, Bn / BLOCK_M);   // 16 x 256 blocks
    dim3 block(NTHREADS);                     // 8 wave32 waves
    hipLaunchKernelGGL(hawkes_lstm_wmma_kernel, grid, block, 0, stream,
                       x, ht, ct, ctar, W, bias, out);
}
